// Transformer_55697135894950
// MI455X (gfx1250) — matrix operand order repaired and checked by reading, not compiled, not run on hardware
//
#include <hip/hip_runtime.h>
#include <hip/hip_bf16.h>

typedef __attribute__((ext_vector_type(16))) __bf16 bf16x16;
typedef __attribute__((ext_vector_type(8)))  __bf16 bf16x8;
typedef __attribute__((ext_vector_type(8)))  float  v8f;
typedef unsigned int u32x4 __attribute__((ext_vector_type(4)));
typedef int          i32x4 __attribute__((ext_vector_type(4)));
typedef int          i32x8 __attribute__((ext_vector_type(8)));

#define BD 2
#define SD 2048
#define CD 1024
#define UD 1024
#define HD 16
#define DH 64
#define HBD (HD*BD)   // 32
#define MD (BD*SD)    // 4096

static __device__ __forceinline__ bf16x16 combine8(const __bf16* p0, const __bf16* p1) {
  bf16x8 lo = *(const bf16x8*)p0;
  bf16x8 hi = *(const bf16x8*)p1;
  return __builtin_shufflevector(lo, hi, 0,1,2,3,4,5,6,7,8,9,10,11,12,13,14,15);
}

static __device__ __forceinline__ v8f wmma_bf16(bf16x16 a, bf16x16 b, v8f c) {
  // D = A(16x32 bf16) * B(32x16 bf16) + C(16x16 f32)
  return __builtin_amdgcn_wmma_f32_16x16x32_bf16(false, a, false, b, (short)0, c, false, false);
}

static __device__ __forceinline__ float redmax16(float v) {
  v = fmaxf(v, __shfl_xor(v, 1, 32));
  v = fmaxf(v, __shfl_xor(v, 2, 32));
  v = fmaxf(v, __shfl_xor(v, 4, 32));
  v = fmaxf(v, __shfl_xor(v, 8, 32));
  return v;
}
static __device__ __forceinline__ float redsum16(float v) {
  v += __shfl_xor(v, 1, 32);
  v += __shfl_xor(v, 2, 32);
  v += __shfl_xor(v, 4, 32);
  v += __shfl_xor(v, 8, 32);
  return v;
}

// ---------------- Tensor Data Mover helpers ----------------
static __device__ __forceinline__ unsigned lds_offset(const void* p) {
  // generic (flat) LDS pointer: low 32 bits are the LDS byte offset
  return (unsigned)(uintptr_t)p;
}

// 2-D TDM load of 2-byte elements: tile_d0 contiguous elems, tile_d1 rows,
// row stride = stride0 elements. D# per CDNA5 ISA ch.8.
static __device__ __forceinline__ void tdm_load_2d_bf16(const void* gptr, unsigned ldsoff,
                                                        int tensor_d0, int tensor_d1,
                                                        int tile_d0, int tile_d1,
                                                        int stride0) {
  unsigned long long ga = (unsigned long long)(uintptr_t)gptr;
  u32x4 g0;
  g0[0] = 1u;                                              // count=1, user mode
  g0[1] = ldsoff;                                          // lds_addr (bytes)
  g0[2] = (unsigned)(ga & 0xffffffffu);                    // global_addr[31:0] -> bits 95:64
  g0[3] = (unsigned)((ga >> 32) & 0x1ffffffu) | (2u << 30);// global_addr[56:32], type=2
  i32x8 g1;
  g1[0] = (1 << 16);                                       // data_size=1 (2 bytes)
  g1[1] = (tensor_d0 & 0xffff) << 16;                      // tensor_dim0 -> bits 79:48
  g1[2] = ((tensor_d0 >> 16) & 0xffff) | ((tensor_d1 & 0xffff) << 16); // dim0 hi | dim1 lo
  g1[3] = ((tensor_d1 >> 16) & 0xffff) | ((tile_d0 & 0xffff) << 16);   // dim1 hi | tile_dim0
  g1[4] = (tile_d1 & 0xffff);                              // tile_dim1 (tile_dim2 = 0)
  g1[5] = stride0;                                         // tensor_dim0_stride[31:0]
  g1[6] = 0;                                               // stride0 hi | stride1 lo
  g1[7] = 0;
  i32x4 z4 = {0, 0, 0, 0};
#if __clang_major__ >= 23
  i32x8 z8 = {0, 0, 0, 0, 0, 0, 0, 0};
  __builtin_amdgcn_tensor_load_to_lds(g0, g1, z4, z4, z8, 0);
#else
  __builtin_amdgcn_tensor_load_to_lds(g0, g1, z4, z4, 0);
#endif
}

static __device__ __forceinline__ void wait_tensor0() {
  __builtin_amdgcn_s_wait_tensorcnt(0);
  asm volatile("" ::: "memory");
}
static __device__ __forceinline__ void wait_tensor2() {
  __builtin_amdgcn_s_wait_tensorcnt(2);
  asm volatile("" ::: "memory");
}
static __device__ __forceinline__ void fence_ds() {
  asm volatile("s_wait_dscnt 0x0" ::: "memory");
}

// ---------------- conversion kernels ----------------
__global__ __launch_bounds__(256) void tx_cvt_x(const float* __restrict__ X,
                                                __bf16* __restrict__ Xb) {
  size_t i = (size_t)blockIdx.x * 256 + threadIdx.x;
  Xb[i] = (__bf16)X[i];
}

__global__ __launch_bounds__(256) void tx_cvt_w(const float* __restrict__ Wq,
                                                const float* __restrict__ Wk,
                                                const float* __restrict__ Wv,
                                                __bf16* __restrict__ Wtq,
                                                __bf16* __restrict__ Wtk,
                                                __bf16* __restrict__ Wtv) {
  size_t i = (size_t)blockIdx.x * 256 + threadIdx.x;   // over C*U
  int k = (int)(i >> 10);          // row of W  (C)
  int n = (int)(i & 1023);         // col of W  (U)
  const float* src = (blockIdx.z == 0) ? Wq : (blockIdx.z == 1) ? Wk : Wv;
  __bf16* dst = (blockIdx.z == 0) ? Wtq : (blockIdx.z == 1) ? Wtk : Wtv;
  dst[(size_t)n * CD + k] = (__bf16)src[i];
}

// ---------------- fused QKV projection (WMMA GEMM) ----------------
// One wave computes a 32(M) x 64(N) tile. K loop in steps of 32 (bf16 WMMA depth).
__global__ __launch_bounds__(32) void tx_qkv_gemm(const __bf16* __restrict__ Xb,
                                                  const __bf16* __restrict__ Wtq,
                                                  const __bf16* __restrict__ Wtk,
                                                  const __bf16* __restrict__ Wtv,
                                                  const float* __restrict__ bq,
                                                  const float* __restrict__ bk,
                                                  const float* __restrict__ bv,
                                                  __bf16* __restrict__ Qb,
                                                  __bf16* __restrict__ Kb,
                                                  __bf16* __restrict__ Vt) {
  const int lane = threadIdx.x & 31;
  const int half = lane >> 4;
  const int lh   = lane & 15;
  const int n0   = blockIdx.x * 64;
  const int m0   = blockIdx.y * 32;
  const int mat  = blockIdx.z;            // 0=Q 1=K 2=V

  const __bf16* Wt  = (mat == 0) ? Wtq : (mat == 1) ? Wtk : Wtv;
  const float* bias = (mat == 0) ? bq  : (mat == 1) ? bk  : bv;

  v8f acc[2][4] = {};

  const int dA0 = half * 16;       // A-frag K offsets: contiguous k = 16*half + s
  const int dA1 = half * 16 + 8;

  for (int kb = 0; kb < CD; kb += 32) {
    if (kb + 32 < CD)
      __builtin_prefetch(Xb + (size_t)(m0 + lh) * CD + kb + 32, 0, 3);
    bf16x16 a[2];
#pragma unroll
    for (int mt = 0; mt < 2; ++mt) {
      const __bf16* rp = Xb + (size_t)(m0 + mt * 16 + lh) * CD + kb;
      a[mt] = combine8(rp + dA0, rp + dA1);
    }
#pragma unroll
    for (int nt = 0; nt < 4; ++nt) {
      const __bf16* bp = Wt + (size_t)(n0 + nt * 16 + lh) * CD + kb + half * 16;
      bf16x16 bm = *(const bf16x16*)bp;
      acc[0][nt] = wmma_bf16(a[0], bm, acc[0][nt]);
      acc[1][nt] = wmma_bf16(a[1], bm, acc[1][nt]);
    }
  }

#pragma unroll
  for (int mt = 0; mt < 2; ++mt) {
#pragma unroll
    for (int nt = 0; nt < 4; ++nt) {
      const int n  = n0 + nt * 16 + lh;
      const int h  = n >> 6;        // head
      const int dl = n & 63;        // d within head
#pragma unroll
      for (int r = 0; r < 8; ++r) {
        const int m = m0 + mt * 16 + r + 8 * half;
        const int b = m >> 11;          // S = 2048
        const int s = m & 2047;
        const int hb = h * BD + b;
        float v = acc[mt][nt][r] + bias[n];
        if (mat == 0) {
          Qb[((size_t)hb * SD + s) * DH + dl] = (__bf16)(v * 0.125f); // 1/sqrt(64)
        } else if (mat == 1) {
          Kb[((size_t)hb * SD + s) * DH + dl] = (__bf16)v;
        } else {
          Vt[((size_t)hb * DH + dl) * SD + s] = (__bf16)v;            // V transposed
        }
      }
    }
  }
}

// ---------------- causal flash attention ----------------
// One wave = 16 queries x dh=64. K/V tiles double-buffered in LDS via TDM.
__global__ __launch_bounds__(32) void tx_attn(const __bf16* __restrict__ Qb,
                                              const __bf16* __restrict__ Kb,
                                              const __bf16* __restrict__ Vt,
                                              const float* __restrict__ X,
                                              float* __restrict__ Out) {
  __shared__ __align__(16) __bf16 Ktile[2][32 * 64];   // [key][d]
  __shared__ __align__(16) __bf16 Vtile[2][64 * 32];   // [d][key]
  __shared__ __align__(16) __bf16 P[16 * 32];
  const int lane = threadIdx.x & 31;
  const int half = lane >> 4;
  const int lh   = lane & 15;
  const int hb   = blockIdx.y;
  const int q0   = blockIdx.x * 16;
  const int h    = hb >> 1;       // hb = h*B + b
  const int b    = hb & 1;

  const __bf16* kgl = Kb + (size_t)hb * SD * DH;   // [S][dh]
  const __bf16* vgl = Vt + (size_t)hb * DH * SD;   // [dh][S]

  // Q A-fragments (queries q0..q0+15, d split into two K=32 chunks)
  bf16x16 aQ[2];
  {
    const __bf16* rp = Qb + ((size_t)hb * SD + q0 + lh) * DH;
#pragma unroll
    for (int t = 0; t < 2; ++t)
      aQ[t] = combine8(rp + t * 32 + half * 16, rp + t * 32 + half * 16 + 8);
  }

  // prime buffer 0 with first K/V tiles (TDM async DMA into LDS)
  tdm_load_2d_bf16(kgl, lds_offset(&Ktile[0][0]), DH, 32, DH, 32, DH);
  tdm_load_2d_bf16(vgl, lds_offset(&Vtile[0][0]), 32, DH, 32, DH, SD);

  v8f o[4] = {};
  float mrow[8], lrow[8];
#pragma unroll
  for (int r = 0; r < 8; ++r) { mrow[r] = -3.0e38f; lrow[r] = 0.0f; }

  int buf = 0;
  for (int kb = 0; kb < q0 + 16; kb += 32) {
    const bool more = (kb + 32 < q0 + 16);
    if (more) {
      fence_ds();   // all prior ds reads of buf^1 retired before TDM overwrites it
      tdm_load_2d_bf16(kgl + (size_t)(kb + 32) * DH,
                       lds_offset(&Ktile[buf ^ 1][0]), DH, 32, DH, 32, DH);
      tdm_load_2d_bf16(vgl + (kb + 32),
                       lds_offset(&Vtile[buf ^ 1][0]), 32, DH, 32, DH, SD);
      wait_tensor2();   // in-order TDM: current pair complete, next pair in flight
    } else {
      wait_tensor0();
    }
    const __bf16* kt = &Ktile[buf][0];
    const __bf16* vt = &Vtile[buf][0];

    // scores: two 16x16 key tiles, each needs K-depth 64 = 2 WMMAs
    v8f sc0 = {}, sc1 = {};
#pragma unroll
    for (int t = 0; t < 2; ++t) {
      bf16x16 b0 = *(const bf16x16*)(kt + (lh)      * DH + t * 32 + half * 16);
      bf16x16 b1 = *(const bf16x16*)(kt + (16 + lh) * DH + t * 32 + half * 16);
      sc0 = wmma_bf16(aQ[t], b0, sc0);
      sc1 = wmma_bf16(aQ[t], b1, sc1);
    }

    const int key0 = kb + lh;
    const int key1 = kb + 16 + lh;
#pragma unroll
    for (int r = 0; r < 8; ++r) {
      const int qrow = q0 + 8 * half + r;
      float s0 = (key0 > qrow) ? -3.0e38f : sc0[r];
      float s1 = (key1 > qrow) ? -3.0e38f : sc1[r];
      float mx = redmax16(fmaxf(s0, s1));
      float mnew = fmaxf(mrow[r], mx);
      float corr = __expf(mrow[r] - mnew);
      mrow[r] = mnew;
      float p0 = __expf(s0 - mnew);
      float p1 = __expf(s1 - mnew);
      float rs = redsum16(p0 + p1);
      lrow[r] = lrow[r] * corr + rs;
      o[0][r] *= corr; o[1][r] *= corr; o[2][r] *= corr; o[3][r] *= corr;
      P[(8 * half + r) * 32 + lh]      = (__bf16)p0;
      P[(8 * half + r) * 32 + 16 + lh] = (__bf16)p1;
    }
    // reload P in A-fragment layout (DS ops are in-order within a wave)
    bf16x16 aP = combine8(&P[lh * 32 + half * 16], &P[lh * 32 + half * 16 + 8]);
#pragma unroll
    for (int t = 0; t < 4; ++t) {
      o[t] = wmma_bf16(aP, *(const bf16x16*)(vt + (t * 16 + lh) * 32 + half * 16), o[t]);
    }
    buf ^= 1;
  }

  // epilogue: normalize, merge heads, add residual
#pragma unroll
  for (int t = 0; t < 4; ++t) {
    const int d = t * 16 + lh;
#pragma unroll
    for (int r = 0; r < 8; ++r) {
      const int row = q0 + r + 8 * half;
      const float val = o[t][r] / lrow[r];
      const size_t idx = ((size_t)b * SD + row) * UD + h * DH + d;
      Out[idx] = val + X[idx];
    }
  }
}

// ---------------- LayerNorm over (S,U) per batch ----------------
__global__ __launch_bounds__(256) void tx_ln_partial(const float* __restrict__ Out,
                                                     float* __restrict__ part) {
  const int bat = blockIdx.y;
  const int blk = blockIdx.x;                       // 128 blocks per batch
  const size_t chunk = (size_t)SD * UD / 128;       // 16384
  const size_t start = (size_t)bat * SD * UD + (size_t)blk * chunk;
  float s = 0.f, ss = 0.f;
  for (size_t i = threadIdx.x; i < chunk; i += 256) {
    float v = Out[start + i];
    s += v; ss += v * v;
  }
  __shared__ float sh[256], sh2[256];
  sh[threadIdx.x] = s; sh2[threadIdx.x] = ss;
  __syncthreads();
  for (int off = 128; off; off >>= 1) {
    if ((int)threadIdx.x < off) {
      sh[threadIdx.x]  += sh[threadIdx.x + off];
      sh2[threadIdx.x] += sh2[threadIdx.x + off];
    }
    __syncthreads();
  }
  if (threadIdx.x == 0) {
    part[(bat * 128 + blk) * 2 + 0] = sh[0];
    part[(bat * 128 + blk) * 2 + 1] = sh2[0];
  }
}

__global__ __launch_bounds__(32) void tx_ln_finalize(const float* __restrict__ part,
                                                     float* __restrict__ stat) {
  const int bat = threadIdx.x;
  if (bat < BD) {
    float s = 0.f, ss = 0.f;
    for (int i = 0; i < 128; ++i) {            // fixed order -> deterministic
      s  += part[(bat * 128 + i) * 2 + 0];
      ss += part[(bat * 128 + i) * 2 + 1];
    }
    const float inv = 1.0f / ((float)SD * (float)UD);
    float mu  = s * inv;
    float var = ss * inv - mu * mu;
    stat[bat * 2 + 0] = mu;
    stat[bat * 2 + 1] = rsqrtf(var + 1e-5f);
  }
}

__global__ __launch_bounds__(256) void tx_ln_apply(float* __restrict__ Out,
                                                   const float* __restrict__ stat,
                                                   const float* __restrict__ gamma,
                                                   const float* __restrict__ beta) {
  size_t idx = (size_t)blockIdx.x * 256 + threadIdx.x;     // over B*S*U
  int bat = (int)(idx >> 21);                              // S*U = 2^21
  size_t su = idx & ((1u << 21) - 1);
  float mu = stat[bat * 2 + 0], rstd = stat[bat * 2 + 1];
  Out[idx] = (Out[idx] - mu) * rstd * gamma[su] + beta[su];
}

// ---------------- launch ----------------
extern "C" void kernel_launch(void* const* d_in, const int* in_sizes, int n_in,
                              void* d_out, int out_size, void* d_ws, size_t ws_size,
                              hipStream_t stream) {
  (void)in_sizes; (void)n_in; (void)out_size; (void)ws_size;
  const float* X     = (const float*)d_in[0];
  const float* Wq    = (const float*)d_in[1];
  const float* bq    = (const float*)d_in[2];
  const float* Wk    = (const float*)d_in[3];
  const float* bk    = (const float*)d_in[4];
  const float* Wv    = (const float*)d_in[5];
  const float* bv    = (const float*)d_in[6];
  const float* gamma = (const float*)d_in[7];
  const float* beta  = (const float*)d_in[8];
  float* Out = (float*)d_out;

  char* ws = (char*)d_ws;
  const size_t MB = 1024u * 1024u;
  __bf16* Xb  = (__bf16*)(ws + 0);          // 8 MiB  : [M][C] bf16
  __bf16* Wtq = (__bf16*)(ws + 8 * MB);     // 2 MiB  : [U][C] bf16
  __bf16* Wtk = (__bf16*)(ws + 10 * MB);
  __bf16* Wtv = (__bf16*)(ws + 12 * MB);
  __bf16* Qb  = (__bf16*)(ws + 14 * MB);    // 8 MiB  : [HB][S][dh]
  __bf16* Kb  = (__bf16*)(ws + 22 * MB);    // 8 MiB  : [HB][S][dh]
  __bf16* Vt  = (__bf16*)(ws + 30 * MB);    // 8 MiB  : [HB][dh][S]
  float*  part = (float*)(ws + 38 * MB);    // 2*128*2 floats
  float*  stat = part + 2 * 128 * 2;        // 4 floats

  tx_cvt_x<<<(MD * CD) / 256, 256, 0, stream>>>(X, Xb);
  tx_cvt_w<<<dim3((CD * UD) / 256, 1, 3), 256, 0, stream>>>(Wq, Wk, Wv, Wtq, Wtk, Wtv);
  tx_qkv_gemm<<<dim3(UD / 64, MD / 32, 3), 32, 0, stream>>>(Xb, Wtq, Wtk, Wtv,
                                                            bq, bk, bv, Qb, Kb, Vt);
  tx_attn<<<dim3(SD / 16, HBD), 32, 0, stream>>>(Qb, Kb, Vt, X, Out);
  tx_ln_partial<<<dim3(128, BD), 256, 0, stream>>>(Out, part);
  tx_ln_finalize<<<1, 32, 0, stream>>>(part, stat);
  tx_ln_apply<<<(BD * SD * UD) / 256, 256, 0, stream>>>(Out, stat, gamma, beta);
}
